// RedNet_68745246540389
// MI455X (gfx1250) — compile-verified, matrix operand order checked by reading, not run on hardware
//
#include <hip/hip_runtime.h>
#include <hip/hip_bf16.h>

// ---------------------------------------------------------------------------
// RedNet (involution ResNet) forward for MI455X / gfx1250.
// 1x1 convs (FLOP-dominant) -> double-buffered LDS-staged wave32 WMMA GEMM
// (v_wmma_f32_16x16x32_f16, f32 accumulate), filling the A tile with CDNA5
// async global->LDS copies (ASYNCcnt) when the builtin is available.
// Elementwise / reduction ops are VALU kernels (HBM-bound at 23.3 TB/s).
// ---------------------------------------------------------------------------

typedef _Float16 v16h __attribute__((ext_vector_type(16)));
typedef _Float16 h8   __attribute__((ext_vector_type(8)));
typedef _Float16 h4   __attribute__((ext_vector_type(4)));
typedef float    v8f  __attribute__((ext_vector_type(8)));
typedef int      v2i  __attribute__((ext_vector_type(2)));

#define LPITCH 40   // LDS tile row pitch in halfs (80B: 16B-aligned b128 reads)

// CDNA5 async global->LDS copy (tracked by ASYNCcnt), if this clang has it.
#if defined(__has_builtin)
#  if __has_builtin(__builtin_amdgcn_global_load_async_to_lds_b64)
#    define USE_ASYNC_LDS 1
#  endif
#endif
#ifndef USE_ASYNC_LDS
#  define USE_ASYNC_LDS 0
#endif

#if USE_ASYNC_LDS
#  define ASYNC_WAIT() asm volatile("s_wait_asynccnt 0" ::: "memory")
#else
#  define ASYNC_WAIT()
#endif

// f32 -> f16 conversion (vectorized), used for W and X before each GEMM.
__global__ __launch_bounds__(256)
void k_cvt_f16(const float* __restrict__ x, _Float16* __restrict__ y, int n)
{
    int i = 4 * (blockIdx.x * blockDim.x + threadIdx.x);
    if (i + 3 < n) {
        float4 v = *(const float4*)(x + i);
        h4 o = { (_Float16)v.x, (_Float16)v.y, (_Float16)v.z, (_Float16)v.w };
        *(h4*)(y + i) = o;
    } else {
        for (int t = i; t < n; ++t) y[t] = (_Float16)x[t];
    }
}

// ---------------------------------------------------------------------------
// WMMA GEMM for 1x1 conv:  Y[b,m,ho,wo] = sum_k W[m,k] * X[b,k,ho*s,wo*s]
// Block: 256 threads (8 waves) -> 64x64 output tile; K-step 32; LDS tiles
// double-buffered (prefetch tile t+1 while computing tile t, 1 barrier/step).
// Fragment layout per CDNA5 ISA 7.12.2: lane L (hi=L>>4) fragment halfs are
// K = hi*8..+7 then 16+hi*8..+7 of row m=L&15 (col n for B) -> 2x ds_load_b128.
// ---------------------------------------------------------------------------
__global__ __launch_bounds__(256)
void k_gemm1x1(const _Float16* __restrict__ W16, const _Float16* __restrict__ X16,
               const float* __restrict__ bias, float* __restrict__ Y,
               int M, int K, int HWin, int Win, int stride,
               int Ho, int Wo, int Batch)
{
    __shared__ __align__(16) _Float16 sA[2][64 * LPITCH];
    __shared__ __align__(16) _Float16 sB[2][64 * LPITCH];

    const int HWo = Ho * Wo;
    const int N   = Batch * HWo;
    const int tid  = threadIdx.x;
    const int lane = tid & 31;
    const int wave = tid >> 5;
    const int l15  = lane & 15;
    const int hi   = lane >> 4;
    const int m0   = blockIdx.y * 64;
    const int j0   = blockIdx.x * 64;

    // ---- A-fill mapping: thread -> rows (ar, ar+32), 4-half k-group akg
    const int ar    = tid >> 3;          // 0..31
    const int akg   = (tid & 7) * 4;     // 0..28
    const int arow0 = m0 + ar;
    const int arow1 = m0 + ar + 32;
    const bool a0ok = arow0 < M;
    const bool a1ok = arow1 < M;
    const _Float16* aptr0 = W16 + (size_t)(a0ok ? arow0 : 0) * K + akg;
    const _Float16* aptr1 = W16 + (size_t)(a1ok ? arow1 : 0) * K + akg;

    // ---- B-fill mapping: thread -> one column n, k = bkb + 4*i (i<8)
    const int bn  = tid & 63;
    const int bkb = tid >> 6;            // 0..3
    const int j   = j0 + bn;
    const bool jok = j < N;
    int bimg = jok ? (j / HWo) : 0;
    int p    = jok ? (j - bimg * HWo) : 0;
    int ho   = p / Wo, wo = p - ho * Wo;
    const _Float16* xbase = X16 + ((size_t)bimg * K) * HWin
                          + (size_t)(ho * stride) * Win + (size_t)(wo * stride);

    // ---- per-wave compute tiles: 4x2 waves, each wave two 16x16 tiles in N
    const int wm = wave & 3;
    const int wn = wave >> 2;
    const int arowf = wm * 16 + l15;
    const int bcol0 = wn * 32 + l15;
    const int bcol1 = bcol0 + 16;

    // ---- tile fill helpers (branchless where it matters)
    auto fillA = [&](int buf, int k0) {
        const bool kv  = (k0 + akg) < K;    // K % 4 == 0 -> whole-group guard
        const bool ok0 = kv && a0ok;
        const bool ok1 = kv && a1ok;
        _Float16* d0 = &sA[buf][ar * LPITCH + akg];
        _Float16* d1 = &sA[buf][(ar + 32) * LPITCH + akg];
        h4 z = {};
#if USE_ASYNC_LDS
        if (ok0)
            __builtin_amdgcn_global_load_async_to_lds_b64(
                (__attribute__((address_space(1))) v2i*)(aptr0 + k0),
                (__attribute__((address_space(3))) v2i*)d0, 0, 0);
        else
            *(h4*)d0 = z;
        if (ok1)
            __builtin_amdgcn_global_load_async_to_lds_b64(
                (__attribute__((address_space(1))) v2i*)(aptr1 + k0),
                (__attribute__((address_space(3))) v2i*)d1, 0, 0);
        else
            *(h4*)d1 = z;
#else
        const _Float16* p0 = ok0 ? (aptr0 + k0) : W16;   // clamped valid addr
        const _Float16* p1 = ok1 ? (aptr1 + k0) : W16;
        h4 v0 = *(const h4*)p0;
        h4 v1 = *(const h4*)p1;
        if (!ok0) v0 = z;
        if (!ok1) v1 = z;
        *(h4*)d0 = v0;
        *(h4*)d1 = v1;
#endif
    };
    auto fillB = [&](int buf, int k0) {
        _Float16* bp = &sB[buf][bn * LPITCH + bkb];
        const _Float16* xp = xbase + (size_t)(k0 + bkb) * HWin;
#pragma unroll
        for (int i = 0; i < 8; ++i) {
            const int kk = k0 + bkb + 4 * i;
            const bool ok = jok && (kk < K);
            const _Float16* sp = ok ? (xp + (size_t)(4 * i) * HWin) : X16;
            _Float16 v = *sp;                         // unconditional load
            bp[4 * i] = ok ? v : (_Float16)0.f;       // select to zero
        }
    };

    v8f acc0 = {}, acc1 = {};

    fillA(0, 0);
    fillB(0, 0);
    ASYNC_WAIT();
    __syncthreads();

    const int nsteps = (K + 31) >> 5;
    for (int t = 0; t < nsteps; ++t) {
        const int cur = t & 1;
        const int nxt = cur ^ 1;
        const int k0n = (t + 1) << 5;
        if (k0n < K) {                 // prefetch next tile while computing
            fillA(nxt, k0n);
            fillB(nxt, k0n);
        }

        const _Float16* Ab = sA[cur];
        const _Float16* Bb = sB[cur];
        h8 al = *(const h8*)&Ab[arowf * LPITCH + hi * 8];
        h8 ah = *(const h8*)&Ab[arowf * LPITCH + 16 + hi * 8];
        v16h a = __builtin_shufflevector(al, ah, 0,1,2,3,4,5,6,7,8,9,10,11,12,13,14,15);
        h8 b0l = *(const h8*)&Bb[bcol0 * LPITCH + hi * 8];
        h8 b0h = *(const h8*)&Bb[bcol0 * LPITCH + 16 + hi * 8];
        v16h b0 = __builtin_shufflevector(b0l, b0h, 0,1,2,3,4,5,6,7,8,9,10,11,12,13,14,15);
        h8 b1l = *(const h8*)&Bb[bcol1 * LPITCH + hi * 8];
        h8 b1h = *(const h8*)&Bb[bcol1 * LPITCH + 16 + hi * 8];
        v16h b1 = __builtin_shufflevector(b1l, b1h, 0,1,2,3,4,5,6,7,8,9,10,11,12,13,14,15);

        acc0 = __builtin_amdgcn_wmma_f32_16x16x32_f16(false, a, false, b0,
                                                      (short)0, acc0, false, false);
        acc1 = __builtin_amdgcn_wmma_f32_16x16x32_f16(false, a, false, b1,
                                                      (short)0, acc1, false, false);

        ASYNC_WAIT();                  // next tile's async fills landed
        __syncthreads();               // + this wave's ds stores / ds reads done
    }

    // ---- store (C/D layout: lane col n=L&15, VGPR v -> row m = v + 8*hi)
    const int mbase = m0 + wm * 16;
    {
        int jj = j0 + wn * 32 + l15;
        if (jj < N) {
            int bi2 = jj / HWo; int pp = jj - bi2 * HWo;
            float* ycol = Y + ((size_t)bi2 * M) * HWo + pp;
#pragma unroll
            for (int v = 0; v < 8; ++v) {
                int m = mbase + v + 8 * hi;
                if (m < M) {
                    float r = acc0[v];
                    if (bias) r += bias[m];
                    ycol[(size_t)m * HWo] = r;
                }
            }
        }
    }
    {
        int jj = j0 + wn * 32 + 16 + l15;
        if (jj < N) {
            int bi2 = jj / HWo; int pp = jj - bi2 * HWo;
            float* ycol = Y + ((size_t)bi2 * M) * HWo + pp;
#pragma unroll
            for (int v = 0; v < 8; ++v) {
                int m = mbase + v + 8 * hi;
                if (m < M) {
                    float r = acc1[v];
                    if (bias) r += bias[m];
                    ycol[(size_t)m * HWo] = r;
                }
            }
        }
    }
}

// ---------------------------------------------------------------------------
// Direct 3x3 conv (stem only), pad=1, optional bias.
// ---------------------------------------------------------------------------
__global__ __launch_bounds__(256)
void k_conv3x3(const float* __restrict__ x, const float* __restrict__ w,
               const float* __restrict__ bias, float* __restrict__ y,
               int B, int Cin, int Hin, int Win, int Cout, int Ho, int Wo, int stride)
{
    int i = blockIdx.x * blockDim.x + threadIdx.x;
    int total = B * Cout * Ho * Wo;
    if (i >= total) return;
    int wo = i % Wo; int t = i / Wo;
    int ho = t % Ho; t /= Ho;
    int co = t % Cout; int b = t / Cout;

    float acc = bias ? bias[co] : 0.f;
    const float* wp = w + (size_t)co * Cin * 9;
    const float* xb = x + (size_t)b * Cin * Hin * Win;
    for (int ci = 0; ci < Cin; ++ci) {
        const float* xc = xb + (size_t)ci * Hin * Win;
#pragma unroll
        for (int kh = 0; kh < 3; ++kh) {
            int h = ho * stride - 1 + kh;
            if ((unsigned)h >= (unsigned)Hin) continue;
#pragma unroll
            for (int kw = 0; kw < 3; ++kw) {
                int wc = wo * stride - 1 + kw;
                if ((unsigned)wc >= (unsigned)Win) continue;
                acc += wp[ci * 9 + kh * 3 + kw] * xc[h * Win + wc];
            }
        }
    }
    y[i] = acc;
}

// ---------------------------------------------------------------------------
// Training-mode BN stats: per-channel mean / population var over (B, H*W).
// ---------------------------------------------------------------------------
__global__ __launch_bounds__(256)
void k_bn_stats(const float* __restrict__ x, float* __restrict__ mean,
                float* __restrict__ var, int B, int C, int HW)
{
    const int c   = blockIdx.x;
    const int tid = threadIdx.x;
    const int n   = B * HW;
    float s = 0.f, s2 = 0.f;
    for (int i = tid; i < n; i += 256) {
        int b = i / HW; int p = i - b * HW;
        float v = x[((size_t)(b * C + c)) * HW + p];
        s += v; s2 += v * v;
    }
    __shared__ float sh[256], sh2[256];
    sh[tid] = s; sh2[tid] = s2;
    __syncthreads();
    for (int st = 128; st > 0; st >>= 1) {
        if (tid < st) { sh[tid] += sh[tid + st]; sh2[tid] += sh2[tid + st]; }
        __syncthreads();
    }
    if (tid == 0) {
        float m = sh[0] / (float)n;
        mean[c] = m;
        var[c]  = sh2[0] / (float)n - m * m;
    }
}

// y = gamma*(x-mean)*rsqrt(var+eps)+beta  [+ res]  [relu]
__global__ __launch_bounds__(256)
void k_bn_apply(const float* __restrict__ x, const float* __restrict__ res,
                const float* __restrict__ mean, const float* __restrict__ var,
                const float* __restrict__ g, const float* __restrict__ bta,
                float* __restrict__ y, int C, int HW, int total, int dorelu)
{
    int i = blockIdx.x * blockDim.x + threadIdx.x;
    if (i >= total) return;
    int c = (i / HW) % C;
    float v = g[c] * (x[i] - mean[c]) * rsqrtf(var[c] + 1e-5f) + bta[c];
    if (res) v += res[i];
    if (dorelu) v = fmaxf(v, 0.f);
    y[i] = v;
}

// non-overlapping s x s average pool
__global__ __launch_bounds__(256)
void k_avgpool(const float* __restrict__ x, float* __restrict__ y,
               int NC, int Hin, int Win, int s, int Ho, int Wo)
{
    int i = blockIdx.x * blockDim.x + threadIdx.x;
    int total = NC * Ho * Wo;
    if (i >= total) return;
    int wo = i % Wo; int t = i / Wo;
    int ho = t % Ho; int nc = t / Ho;
    const float* xp = x + ((size_t)nc * Hin + (size_t)ho * s) * Win + (size_t)wo * s;
    float acc = 0.f;
    for (int ih = 0; ih < s; ++ih)
        for (int iw = 0; iw < s; ++iw)
            acc += xp[ih * Win + iw];
    y[i] = acc / (float)(s * s);
}

// 3x3 stride-2 max pool, pad 1
__global__ __launch_bounds__(256)
void k_maxpool(const float* __restrict__ x, float* __restrict__ y,
               int NC, int Hin, int Win, int Ho, int Wo)
{
    int i = blockIdx.x * blockDim.x + threadIdx.x;
    int total = NC * Ho * Wo;
    if (i >= total) return;
    int wo = i % Wo; int t = i / Wo;
    int ho = t % Ho; int nc = t / Ho;
    const float* xp = x + (size_t)nc * Hin * Win;
    float m = -3.402823466e+38f;
#pragma unroll
    for (int kh = 0; kh < 3; ++kh) {
        int h = ho * 2 - 1 + kh;
        if ((unsigned)h >= (unsigned)Hin) continue;
#pragma unroll
        for (int kw = 0; kw < 3; ++kw) {
            int wc = wo * 2 - 1 + kw;
            if ((unsigned)wc >= (unsigned)Win) continue;
            m = fmaxf(m, xp[h * Win + wc]);
        }
    }
    y[i] = m;
}

// Involution: out[b,c,ho,wo] = sum_{kh,kw} wk[b, g*9+kh*3+kw, ho, wo] *
//                                          x[b, c, ho*s-1+kh, wo*s-1+kw]
__global__ __launch_bounds__(256)
void k_involution(const float* __restrict__ x, const float* __restrict__ wk,
                  float* __restrict__ y, int B, int C, int G,
                  int Hin, int Win, int stride, int Ho, int Wo)
{
    int i = blockIdx.x * blockDim.x + threadIdx.x;
    int total = B * C * Ho * Wo;
    if (i >= total) return;
    int wo = i % Wo; int t = i / Wo;
    int ho = t % Ho; t /= Ho;
    int c = t % C; int b = t / C;
    int cg = C / G;
    int g = c / cg;
    const float* xp = x + ((size_t)(b * C + c)) * Hin * Win;
    const float* wp = wk + ((size_t)(b * G + g) * 9) * Ho * Wo + (size_t)ho * Wo + wo;
    float acc = 0.f;
#pragma unroll
    for (int kh = 0; kh < 3; ++kh) {
        int h = ho * stride - 1 + kh;
        if ((unsigned)h >= (unsigned)Hin) continue;
#pragma unroll
        for (int kw = 0; kw < 3; ++kw) {
            int wc = wo * stride - 1 + kw;
            if ((unsigned)wc >= (unsigned)Win) continue;
            acc += wp[(size_t)(kh * 3 + kw) * Ho * Wo] * xp[h * Win + wc];
        }
    }
    y[i] = acc;
}

// ---------------------------------------------------------------------------
// Host orchestration. Input flattening assumed in dict insertion order:
//   [0]=x, stem: c1_w,c1_b, r_w,r_g,r_b,s_w,s_b, bn_g,bn_b, c2_w,c2_b (1..11)
//   then 16 blocks: c1_w,n1_g,n1_b, r_w,r_g,r_b,s_w,s_b, n2_g,n2_b,
//                   c3_w,n3_g,n3_b [, ds_w]   (13 or 14 tensors each).
// ---------------------------------------------------------------------------
extern "C" void kernel_launch(void* const* d_in, const int* in_sizes, int n_in,
                              void* d_out, int out_size, void* d_ws, size_t ws_size,
                              hipStream_t stream)
{
    (void)in_sizes; (void)n_in; (void)ws_size;
    const int B = 8;
    auto F = [&](int i) { return (const float*)d_in[i]; };

    // --- workspace carve (float units; all chunks multiples of 4 -> 16B align)
    float* ws = (float*)d_ws;
    size_t off = 0;
    auto wsalloc = [&](size_t n) { float* p = ws + off; off += n; return p; };
    const size_t BIG = 6422528;              // 8*64*112*112 == 8*256*56*56
    float* bufs[4];
    for (int i = 0; i < 4; ++i) bufs[i] = wsalloc(BIG);
    float* tsmall = wsalloc(1 << 20);        // involution reduce-conv act
    float* wkbuf  = wsalloc(28901376);       // 8*288*112*112 stem inv kernels
    float* meanb  = wsalloc(2048);
    float* varb   = wsalloc(2048);
    _Float16* xhalf = (_Float16*)wsalloc(BIG / 2 + 16);   // f16 activations
    _Float16* whalf = (_Float16*)wsalloc(1048592);        // f16 weights (<=2M halfs)

    auto gemm = [&](const float* W, const float* X, const float* bias, float* Y,
                    int M, int K, int Hin, int Win, int stride, int Ho, int Wo) {
        int nW = M * K;
        int nX = B * K * Hin * Win;
        k_cvt_f16<<<dim3((nW / 4 + 255) / 256), 256, 0, stream>>>(W, whalf, nW);
        k_cvt_f16<<<dim3((nX / 4 + 255) / 256), 256, 0, stream>>>(X, xhalf, nX);
        int N = B * Ho * Wo;
        dim3 grid((N + 63) / 64, (M + 63) / 64);
        k_gemm1x1<<<grid, 256, 0, stream>>>(whalf, xhalf, bias, Y, M, K,
                                            Hin * Win, Win, stride, Ho, Wo, B);
    };
    auto bn = [&](const float* x, const float* res, const float* g, const float* bb,
                  float* y, int C, int HW, int dorelu) {
        k_bn_stats<<<dim3(C), 256, 0, stream>>>(x, meanb, varb, B, C, HW);
        int total = B * C * HW;
        k_bn_apply<<<dim3((total + 255) / 256), 256, 0, stream>>>(
            x, res, meanb, varb, g, bb, y, C, HW, total, dorelu);
    };
    auto involution = [&](const float* x, int C, int Hin, int Win, int stride, int G,
                          const float* rW, const float* rG, const float* rB,
                          const float* sW, const float* sB,
                          float* y, float* xsbuf) {
        int Ho = (Hin - 1) / stride + 1, Wo = (Win - 1) / stride + 1;
        int cr = C / 4;
        const float* xs = x;
        if (stride > 1) {
            int total = B * C * Ho * Wo;
            k_avgpool<<<dim3((total + 255) / 256), 256, 0, stream>>>(
                x, xsbuf, B * C, Hin, Win, stride, Ho, Wo);
            xs = xsbuf;
        }
        gemm(rW, xs, nullptr, tsmall, cr, C, Ho, Wo, 1, Ho, Wo);    // reduce 1x1
        bn(tsmall, nullptr, rG, rB, tsmall, cr, Ho * Wo, 1);        // BN+ReLU
        gemm(sW, tsmall, sB, wkbuf, G * 9, cr, Ho, Wo, 1, Ho, Wo);  // span + bias
        int total = B * C * Ho * Wo;
        k_involution<<<dim3((total + 255) / 256), 256, 0, stream>>>(
            x, wkbuf, y, B, C, G, Hin, Win, stride, Ho, Wo);
    };

    // ---------------- stem ----------------
    {
        int total = B * 32 * 112 * 112;
        k_conv3x3<<<dim3((total + 255) / 256), 256, 0, stream>>>(
            F(0), F(1), F(2), bufs[0], B, 3, 224, 224, 32, 112, 112, 2);
    }
    involution(bufs[0], 32, 112, 112, 1, 32, F(3), F(4), F(5), F(6), F(7),
               bufs[1], bufs[2]);
    bn(bufs[1], nullptr, F(8), F(9), bufs[1], 32, 112 * 112, 1);
    {
        int total = B * 64 * 112 * 112;
        k_conv3x3<<<dim3((total + 255) / 256), 256, 0, stream>>>(
            bufs[1], F(10), F(11), bufs[0], B, 32, 112, 112, 64, 112, 112, 1);
    }
    {
        int total = B * 64 * 56 * 56;
        k_maxpool<<<dim3((total + 255) / 256), 256, 0, stream>>>(
            bufs[0], bufs[1], B * 64, 112, 112, 56, 56);
    }

    // ---------------- stages ----------------
    int ci = 1;
    int H = 56;
    int cin = 64;
    int cout = 256;
    int idx = 12;
    const int stage_blocks[4] = {3, 4, 6, 3};
    const int strides_[4]     = {1, 2, 2, 2};

    for (int li = 0; li < 4; ++li) {
        for (int bi = 0; bi < stage_blocks[li]; ++bi) {
            int stride = (bi == 0) ? strides_[li] : 1;
            bool hasds = (bi == 0);
            const float* P[14];
            for (int j = 0; j < 13; ++j) P[j] = F(idx + j);
            if (hasds) P[13] = F(idx + 13);
            idx += hasds ? 14 : 13;

            int o[3], t2 = 0;
            for (int i2 = 0; i2 < 4; ++i2) if (i2 != ci) o[t2++] = i2;
            float* s1 = bufs[o[0]];
            float* s2 = bufs[o[1]];
            float* s3 = bufs[o[2]];
            const float* cur = bufs[ci];

            int m  = cout / 4;
            int Ho = (H - 1) / stride + 1;

            // c1 (1x1) + BN + ReLU at full H
            gemm(P[0], cur, nullptr, s1, m, cin, H, H, 1, H, H);
            bn(s1, nullptr, P[1], P[2], s1, m, H * H, 1);
            // involution (groups=2), stride
            involution(s1, m, H, H, stride, 2, P[3], P[4], P[5], P[6], P[7], s2, s3);
            bn(s2, nullptr, P[8], P[9], s2, m, Ho * Ho, 1);
            // c3 (1x1), then BN + residual + ReLU
            gemm(P[10], s2, nullptr, s1, cout, m, Ho, Ho, 1, Ho, Ho);
            const float* idn = cur;
            if (hasds) {
                gemm(P[13], cur, nullptr, s3, cout, cin, H, H, stride, Ho, Ho);
                idn = s3;
            }
            bn(s1, idn, P[11], P[12], s2, cout, Ho * Ho, 1);

            ci = o[1];
            cin = cout;
            H = Ho;
        }
        cout *= 2;
    }

    (void)hipMemcpyAsync(d_out, bufs[ci], (size_t)out_size * sizeof(float),
                         hipMemcpyDeviceToDevice, stream);
}
